// AntiBurst_49581102465394
// MI455X (gfx1250) — matrix operand order repaired and checked by reading, not compiled, not run on hardware
//
#include <hip/hip_runtime.h>
#include <math.h>

typedef __attribute__((ext_vector_type(16))) _Float16 v16h;
typedef __attribute__((ext_vector_type(8)))  _Float16 v8h;
typedef __attribute__((ext_vector_type(8)))  float    v8f;

#define B_ 32
#define D_ 768
#define N_ 1024

// ---------------------------------------------------------------------------
// Kernel 0: per-batch transpose + f32->f16 convert.
//   x  : (B, D, N) f32, n contiguous
//   xt : (B, N, D) f16, d contiguous   (rows of X^T; serves both WMMA A and B)
// ---------------------------------------------------------------------------
__global__ __launch_bounds__(256) void transpose_cvt_kernel(
    const float* __restrict__ x, _Float16* __restrict__ xt) {
  __shared__ float tile[32][33];
  const int b  = blockIdx.z;
  const int n0 = blockIdx.x * 32;
  const int d0 = blockIdx.y * 32;
  const float* xb  = x  + (size_t)b * D_ * N_;
  _Float16*    xtb = xt + (size_t)b * N_ * D_;
  const int tx = threadIdx.x;   // 0..31
  const int ty = threadIdx.y;   // 0..7
#pragma unroll
  for (int r = 0; r < 4; ++r) {
    int d = ty + 8 * r;
    tile[d][tx] = xb[(size_t)(d0 + d) * N_ + (n0 + tx)];   // coalesced in n
  }
  __syncthreads();
#pragma unroll
  for (int r = 0; r < 4; ++r) {
    int n = ty + 8 * r;
    xtb[(size_t)(n0 + n) * D_ + (d0 + tx)] = (_Float16)tile[tx][n]; // coalesced in d
  }
}

// ---------------------------------------------------------------------------
// Kernel 1: fused Gram + affine + sigmoid + row-sum + pow(-p).
// One wave per 16-row n-tile. Processes 4 m-tiles per K-sweep so the A
// fragment load is shared across 4 WMMAs (10 b128 loads per 4 wmma instead
// of 16). Sigmoid uses v_rcp_f32 (trans op, co-executes with the XDL pipe)
// instead of the IEEE divide sequence. Never materializes the Gram matrix.
// C layout (wave32): VGPR j, lanes 0-15 -> row j, col=lane;
//                    lanes 16-31 -> row j+8, col=lane-16.
// ---------------------------------------------------------------------------
__global__ __launch_bounds__(256) void gram_sigmoid_kernel(
    const _Float16* __restrict__ xt, const float* __restrict__ ab,
    float* __restrict__ winv) {
  const int b    = blockIdx.y;
  const int wave = threadIdx.x >> 5;
  const int lane = threadIdx.x & 31;
  const int lo   = lane & 15;   // row (A) / col (B) within tile
  const int hi   = lane >> 4;   // half-wave selector
  const int n0   = (blockIdx.x * 8 + wave) * 16;

  const _Float16* xtb = xt + (size_t)b * N_ * D_;
  const _Float16* pa  = xtb + (size_t)(n0 + lo) * D_;   // A row n0+lo, contiguous k

  // dis = (-2 + 2g)*ab_w + ab_b = (2*ab_w)*g + (ab_b - 2*ab_w)
  const float sw = 2.0f * ab[0];
  const float sb = ab[1] - 2.0f * ab[0];
  const float p  = ab[2];

  float ssum[8];
#pragma unroll
  for (int j = 0; j < 8; ++j) ssum[j] = 0.0f;

  for (int m0 = 0; m0 < N_; m0 += 64) {       // 4 m-tiles per pass
    const _Float16* pb0 = xtb + (size_t)(m0 +  0 + lo) * D_ + 16 * hi;
    const _Float16* pb1 = xtb + (size_t)(m0 + 16 + lo) * D_ + 16 * hi;
    const _Float16* pb2 = xtb + (size_t)(m0 + 32 + lo) * D_ + 16 * hi;
    const _Float16* pb3 = xtb + (size_t)(m0 + 48 + lo) * D_ + 16 * hi;
    v8f c0 = {}, c1 = {}, c2 = {}, c3 = {};
#pragma unroll 2
    for (int k0 = 0; k0 < D_; k0 += 32) {
      // A 16x32 f16 layout: lane(hi) holds K=[k0+16hi, +16) for its row
      v8h a0 = *(const v8h*)(pa + k0 + 16 * hi);
      v8h a1 = *(const v8h*)(pa + k0 + 16 * hi + 8);
      v16h A;
#pragma unroll
      for (int i = 0; i < 8; ++i) { A[i] = a0[i]; A[i + 8] = a1[i]; }

      // B 32x16 f16 layout: lane(hi) holds K=[k0+16hi, +16) for its column
#define DO_TILE(PB, CC)                                                        \
      {                                                                        \
        v8h b0 = *(const v8h*)((PB) + k0);                                     \
        v8h b1 = *(const v8h*)((PB) + k0 + 8);                                 \
        v16h Bm;                                                               \
        _Pragma("unroll")                                                      \
        for (int i = 0; i < 8; ++i) { Bm[i] = b0[i]; Bm[i + 8] = b1[i]; }      \
        CC = __builtin_amdgcn_wmma_f32_16x16x32_f16(                           \
            false, A, false, Bm, (short)0, CC, false, false);                  \
      }
      DO_TILE(pb0, c0)
      DO_TILE(pb1, c1)
      DO_TILE(pb2, c2)
      DO_TILE(pb3, c3)
#undef DO_TILE
    }
    // sigmoid(sw*g + sb) accumulated into per-row sums (fast rcp: 1 trans op)
#pragma unroll
    for (int j = 0; j < 8; ++j) {
      ssum[j] += __builtin_amdgcn_rcpf(1.0f + __expf(-(sw * c0[j] + sb)));
      ssum[j] += __builtin_amdgcn_rcpf(1.0f + __expf(-(sw * c1[j] + sb)));
      ssum[j] += __builtin_amdgcn_rcpf(1.0f + __expf(-(sw * c2[j] + sb)));
      ssum[j] += __builtin_amdgcn_rcpf(1.0f + __expf(-(sw * c3[j] + sb)));
    }
  }

  // Reduce over the 16 columns held across each 16-lane half.
#pragma unroll
  for (int j = 0; j < 8; ++j) {
    float s = ssum[j];
    s += __shfl_xor(s, 1, 32);
    s += __shfl_xor(s, 2, 32);
    s += __shfl_xor(s, 4, 32);
    s += __shfl_xor(s, 8, 32);
    ssum[j] = s;
  }

  if (lo == 0) {
    // hi=0 lane holds rows n0+0..7 ; hi=1 lane holds rows n0+8..15
    float* wrow = winv + (size_t)b * N_ + n0 + 8 * hi;
#pragma unroll
    for (int j = 0; j < 8; ++j)
      wrow[j] = __powf(ssum[j], -p);           // reciprocal weight w^-p
  }
}

// ---------------------------------------------------------------------------
// Kernel 2: out[b,d,n] = x[b,d,n] * winv[b,n]  (float4 streaming)
// ---------------------------------------------------------------------------
__global__ __launch_bounds__(256) void scale_kernel(
    const float* __restrict__ x, const float* __restrict__ winv,
    float* __restrict__ out) {
  const size_t i4   = (size_t)blockIdx.x * 256 + threadIdx.x; // float4 index
  const size_t base = i4 * 4;
  const int    n    = (int)(base & (N_ - 1));
  const int    b    = (int)(base / ((size_t)D_ * N_));
  const float4 xv = ((const float4*)x)[i4];
  const float4 wv = *(const float4*)(winv + (size_t)b * N_ + n);
  float4 o;
  o.x = xv.x * wv.x;
  o.y = xv.y * wv.y;
  o.z = xv.z * wv.z;
  o.w = xv.w * wv.w;
  ((float4*)out)[i4] = o;
}

// ---------------------------------------------------------------------------
extern "C" void kernel_launch(void* const* d_in, const int* in_sizes, int n_in,
                              void* d_out, int out_size, void* d_ws, size_t ws_size,
                              hipStream_t stream) {
  const float* x  = (const float*)d_in[0];   // (B,D,H,W) f32
  const float* ab = (const float*)d_in[1];   // [ab_w, ab_b, ab_p]
  float* out = (float*)d_out;

  // Workspace: winv (B*N f32 = 128 KB) at 0; XhT (B*N*D f16 = 48 MB) at +1 MB.
  float*    winv = (float*)d_ws;
  _Float16* xt   = (_Float16*)((char*)d_ws + (1u << 20));

  dim3 g0(N_ / 32, D_ / 32, B_);
  dim3 b0(32, 8);
  transpose_cvt_kernel<<<g0, b0, 0, stream>>>(x, xt);

  dim3 g1(N_ / 128, B_);   // 8 waves/block, 16 rows/wave -> 128 rows/block
  gram_sigmoid_kernel<<<g1, 256, 0, stream>>>(xt, ab, winv);

  const size_t total4 = (size_t)B_ * D_ * N_ / 4;
  scale_kernel<<<(unsigned)((total4 + 255) / 256), 256, 0, stream>>>(x, winv, out);
}
